// FlatModel_3521873183179
// MI455X (gfx1250) — matrix operand order repaired and checked by reading, not compiled, not run on hardware
//
#include <hip/hip_runtime.h>
#include <hip/hip_bf16.h>
#include <stdint.h>
#include <stddef.h>

typedef __attribute__((ext_vector_type(16))) _Float16 v16h;
typedef __attribute__((ext_vector_type(8)))  _Float16 v8h;
typedef __attribute__((ext_vector_type(8)))  float    v8f;

#define HID   128
#define N_CLS 10

// Relaxed device-scope f32 atomic add -> should lower to global_atomic_add_f32
// (native on gfx1250, ISA 15.18.3 op 86) instead of a CAS loop.
__device__ __forceinline__ void fadd_atomic(float* p, float v) {
    __hip_atomic_fetch_add(p, v, __ATOMIC_RELAXED, __HIP_MEMORY_SCOPE_AGENT);
}

// ---------------- elementwise / setup kernels ----------------

__global__ void zero_f32(float* __restrict__ p, int n) {
    int i = blockIdx.x * blockDim.x + threadIdx.x;
    if (i < n) p[i] = 0.0f;
}

__global__ void deg_kernel(const long long* __restrict__ dst, float* __restrict__ deg, int E) {
    int e = blockIdx.x * blockDim.x + threadIdx.x;
    if (e < E) fadd_atomic(&deg[(int)dst[e]], 1.0f);
}

// dinv[i] = rsqrt(deg_with_selfloop) ; deg buffer holds edge-only indegree
__global__ void dinv_kernel(float* __restrict__ d, int n) {
    int i = blockIdx.x * blockDim.x + threadIdx.x;
    if (i < n) d[i] = rsqrtf(d[i] + 1.0f);
}

// Pack a 128x128 f32 row-major weight matrix into WMMA B-fragment order (f16).
// Layout index: (((t*4 + kk)*32 + lane)*16 + j)
//   t  = 16-col tile (0..7), kk = K-step of 32 (0..3), lane = 0..31, j = 0..15
//   B element: row = kk*32 + (lane>=16 ? 8:0) + (j&7) + (j>=8 ? 16:0) ; col = t*16 + (lane&15)
__global__ void pack_w(const float* __restrict__ W, _Float16* __restrict__ out) {
    int gid = blockIdx.x * blockDim.x + threadIdx.x;   // 0..16383
    int j    = gid & 15;
    int lane = (gid >> 4) & 31;
    int kk   = (gid >> 9) & 3;
    int t    = gid >> 11;
    int row = kk * 32 + ((lane & 16) ? 8 : 0) + (j & 7) + ((j & 8) ? 16 : 0);
    int col = t * 16 + (lane & 15);
    out[gid] = (_Float16)W[row * HID + col];
}

// ---------------- dual WMMA GEMM: H = X@W ; Agg = X@Ws + b + bs ----------------
// One block = 16 rows of X (N must be a multiple of 16; 50000 = 3125*16).
// 8 waves; wave w owns output columns [16w, 16w+16) for BOTH weight matrices.

__global__ __launch_bounds__(256) void gemm_dual(
    const float* __restrict__ X,
    const _Float16* __restrict__ Wp, const _Float16* __restrict__ Wsp,
    const float* __restrict__ b, const float* __restrict__ bs,
    float* __restrict__ H, float* __restrict__ Agg, int N)
{
    __shared__ __align__(16) _Float16 xa[16][136];   // 16 rows x 128 cols, padded
    const int tid  = threadIdx.x;
    const int row0 = blockIdx.x * 16;
    if (row0 >= N) return;                           // uniform across block

    // stage 16x128 f32 -> f16 in LDS (coalesced)
#pragma unroll
    for (int i = 0; i < 8; ++i) {
        int idx = tid + i * 256;                     // 0..2047
        int r = idx >> 7, c = idx & 127;
        xa[r][c] = (_Float16)X[(size_t)(row0 + r) * HID + c];
    }
    __syncthreads();

    const int w    = tid >> 5;
    const int lane = tid & 31;
    const int m    = lane & 15;                      // A row this lane owns
    const int cb   = (lane & 16) ? 8 : 0;            // A K-offset per ISA layout

    v8f c0 = {};   // X @ W
    v8f c1 = {};   // X @ Ws

#pragma unroll
    for (int kk = 0; kk < 4; ++kk) {
        // A fragment from LDS: a[0..7]=A[m][k0..k0+7], a[8..15]=A[m][k0+16..k0+23]
        const v8h lo = *(const v8h*)&xa[m][kk * 32 + cb];
        const v8h hi = *(const v8h*)&xa[m][kk * 32 + 16 + cb];
        v16h a;
#pragma unroll
        for (int j = 0; j < 8; ++j) { a[j] = lo[j]; a[8 + j] = hi[j]; }

        // B fragments: contiguous 16 halves per lane (pre-packed)
        const size_t boff = (size_t)(((w * 4 + kk) * 32 + lane)) * 16;
        v16h bw  = *(const v16h*)(Wp  + boff);
        v16h bws = *(const v16h*)(Wsp + boff);

        c0 = __builtin_amdgcn_wmma_f32_16x16x32_f16(false, a, false, bw,
                                                    (short)0, c0, false, false);
        c1 = __builtin_amdgcn_wmma_f32_16x16x32_f16(false, a, false, bws,
                                                    (short)0, c1, false, false);
    }

    // store: C/D layout — VGPR r holds row (r + (lane<16?0:8)), col = lane%16
    const int col  = w * 16 + m;
    const int m0   = (lane & 16) ? 8 : 0;
    const float bb = b[col] + bs[col];
#pragma unroll
    for (int r = 0; r < 8; ++r) {
        size_t o = (size_t)(row0 + m0 + r) * HID + col;
        H[o]   = c0[r];
        Agg[o] = c1[r] + bb;                         // seed: self-transform + biases
    }
}

// ---------------- edge scatter: Agg[dst] += dinv[s]*dinv[d]*H[src] ----------------
// One thread = one float4 (32 threads per edge). Gather is global_load_b128;
// 4 non-returning f32 atomics per thread.

__global__ void scatter_kernel(const long long* __restrict__ src,
                               const long long* __restrict__ dst,
                               const float* __restrict__ dinv,
                               const float* __restrict__ H,
                               float* __restrict__ Agg, int E)
{
    unsigned idx = blockIdx.x * blockDim.x + threadIdx.x;
    if (idx >= (unsigned)E * 32u) return;
    int e = idx >> 5, q = idx & 31;                  // q: float4 index within row
    int s = (int)src[e], d = (int)dst[e];
    float wgt = dinv[s] * dinv[d];
    const float4 h = *(const float4*)(H + (size_t)s * HID + q * 4);
    float* a = Agg + (size_t)d * HID + q * 4;
    fadd_atomic(a + 0, wgt * h.x);
    fadd_atomic(a + 1, wgt * h.y);
    fadd_atomic(a + 2, wgt * h.z);
    fadd_atomic(a + 3, wgt * h.w);
}

// ---------------- combine: Y = relu(Agg + dinv^2 * H) ----------------

__global__ void combine_kernel(const float* __restrict__ H,
                               const float* __restrict__ Agg,
                               const float* __restrict__ dinv,
                               float* __restrict__ Y, int N)
{
    unsigned idx = blockIdx.x * blockDim.x + threadIdx.x;
    if (idx >= (unsigned)N * 128u) return;
    int i = idx >> 7;
    float di = dinv[i];
    float v = Agg[idx] + di * di * H[idx];
    Y[idx] = fmaxf(v, 0.0f);
}

// ---------------- mean pool (column sums) + classifier head ----------------

__global__ __launch_bounds__(128) void colsum_kernel(const float* __restrict__ Y,
                                                     float* __restrict__ colsum, int N)
{
    int f  = threadIdx.x;                            // 0..127
    int r0 = blockIdx.x * 256;
    float acc = 0.0f;
    for (int r = 0; r < 256; ++r) {
        int row = r0 + r;
        if (row < N) acc += Y[(size_t)row * HID + f];
    }
    fadd_atomic(&colsum[f], acc);
}

__global__ void head_kernel(const float* __restrict__ colsum,
                            const float* __restrict__ Wc,
                            const float* __restrict__ bc,
                            float* __restrict__ out, int N)
{
    int c = threadIdx.x;
    if (c < N_CLS) {
        float invN = 1.0f / (float)N;
        float acc = 0.0f;
        for (int f = 0; f < HID; ++f)
            acc += colsum[f] * invN * Wc[f * N_CLS + c];
        out[c] = acc + bc[c];
    }
}

// ---------------- launcher ----------------

extern "C" void kernel_launch(void* const* d_in, const int* in_sizes, int n_in,
                              void* d_out, int out_size, void* d_ws, size_t ws_size,
                              hipStream_t stream)
{
    const float*     x   = (const float*)d_in[0];
    const long long* ei  = (const long long*)d_in[1];   // int64 [2, E]
    const float*     W1  = (const float*)d_in[2];
    const float*     b1  = (const float*)d_in[3];
    const float*     Ws1 = (const float*)d_in[4];
    const float*     bs1 = (const float*)d_in[5];
    const float*     W2  = (const float*)d_in[6];
    const float*     b2  = (const float*)d_in[7];
    const float*     Ws2 = (const float*)d_in[8];
    const float*     bs2 = (const float*)d_in[9];
    const float*     Wc  = (const float*)d_in[10];
    const float*     bc  = (const float*)d_in[11];
    float* out = (float*)d_out;

    const int N = in_sizes[0] / HID;      // 50000
    const int E = in_sizes[1] / 2;        // 800000
    const long long* src = ei;
    const long long* dst = ei + E;

    char* ws = (char*)d_ws;
    size_t off = 0;
    auto alloc = [&](size_t bytes) -> void* {
        void* p = ws + off;
        off = (off + bytes + 255) & ~(size_t)255;
        return p;
    };
    float* dinv   = (float*)alloc((size_t)(N + 128) * 4);   // deg->dinv, then colsum
    float* colsum = dinv + N;
    float* hbuf   = (float*)alloc((size_t)N * HID * 4);
    float* aggbuf = (float*)alloc((size_t)N * HID * 4);
    float* ybuf   = (float*)alloc((size_t)N * HID * 4);
    _Float16* W1p  = (_Float16*)alloc(16384 * 2);
    _Float16* Ws1p = (_Float16*)alloc(16384 * 2);
    _Float16* W2p  = (_Float16*)alloc(16384 * 2);
    _Float16* Ws2p = (_Float16*)alloc(16384 * 2);

    const int T = 256;
    unsigned nf    = (unsigned)N * 128u;
    unsigned e4    = (unsigned)E * 32u;                 // float4-granular scatter
    unsigned gRowT = (unsigned)(N / 16);                // 3125 (N is multiple of 16)

    // degree / dinv (re-zeroed every call for graph-replay determinism)
    zero_f32<<<(N + 128 + T - 1) / T, T, 0, stream>>>(dinv, N + 128);
    deg_kernel<<<(E + T - 1) / T, T, 0, stream>>>(dst, dinv, E);
    dinv_kernel<<<(N + T - 1) / T, T, 0, stream>>>(dinv, N);

    // pack weights to WMMA fragment layout (f16)
    pack_w<<<64, T, 0, stream>>>(W1,  W1p);
    pack_w<<<64, T, 0, stream>>>(Ws1, Ws1p);
    pack_w<<<64, T, 0, stream>>>(W2,  W2p);
    pack_w<<<64, T, 0, stream>>>(Ws2, Ws2p);

    // ---- layer 1 ----
    gemm_dual<<<gRowT, T, 0, stream>>>(x, W1p, Ws1p, b1, bs1, hbuf, aggbuf, N);
    scatter_kernel<<<(e4 + T - 1) / T, T, 0, stream>>>(src, dst, dinv, hbuf, aggbuf, E);
    combine_kernel<<<(nf + T - 1) / T, T, 0, stream>>>(hbuf, aggbuf, dinv, ybuf, N);

    // ---- layer 2 ----
    gemm_dual<<<gRowT, T, 0, stream>>>(ybuf, W2p, Ws2p, b2, bs2, hbuf, aggbuf, N);
    scatter_kernel<<<(e4 + T - 1) / T, T, 0, stream>>>(src, dst, dinv, hbuf, aggbuf, E);
    combine_kernel<<<(nf + T - 1) / T, T, 0, stream>>>(hbuf, aggbuf, dinv, ybuf, N);

    // ---- mean pool + head ----
    colsum_kernel<<<(N + 255) / 256, 128, 0, stream>>>(ybuf, colsum, N);
    head_kernel<<<1, 32, 0, stream>>>(colsum, Wc, bc, out, N);
}
